// SelfAttention_v2_63161789055035
// MI455X (gfx1250) — matrix operand order repaired and checked by reading, not compiled, not run on hardware
//
#include <hip/hip_runtime.h>
#include <math.h>

// ---------- types ----------
typedef __attribute__((ext_vector_type(16))) __bf16 v16bf;
typedef __attribute__((ext_vector_type(8)))  float  v8f;
typedef __attribute__((ext_vector_type(4)))  int    v4i;

// pointer-to-addrspace(1)/(3) v4i (attribute qualifies the pointee)
typedef __attribute__((address_space(1))) v4i* gas_v4i_ptr;
typedef __attribute__((address_space(3))) v4i* las_v4i_ptr;

#define SEQ  4096
#define EMB  1024
#define DOUT 1024

// GEMM tiling
#define BM   128      // block rows
#define BN   128      // block cols
#define KB   32       // k-block staged in LDS
#define LST  40       // LDS row stride in bf16 elements (KB + 8 pad -> 80B, conflict-free)

// ---------- async global->LDS support (CDNA5 ASYNCcnt path) ----------
#if defined(__has_builtin)
#if __has_builtin(__builtin_amdgcn_global_load_async_to_lds_b128)
#define HAVE_ASYNC_LDS 1
#endif
#endif
#ifndef HAVE_ASYNC_LDS
#define HAVE_ASYNC_LDS 0
#endif

__device__ __forceinline__ void cp16_g2l(unsigned short* l, const unsigned short* g) {
#if HAVE_ASYNC_LDS
  __builtin_amdgcn_global_load_async_to_lds_b128(
      (gas_v4i_ptr)(unsigned short*)g,   // global src (const dropped; param is non-const)
      (las_v4i_ptr)l,                    // LDS dst
      /*offset=*/0, /*cpol=*/0);
#else
  *reinterpret_cast<uint4*>(l) = *reinterpret_cast<const uint4*>(g);
#endif
}

__device__ __forceinline__ void wait_async_copies() {
#if HAVE_ASYNC_LDS
#if __has_builtin(__builtin_amdgcn_s_wait_asynccnt)
  __builtin_amdgcn_s_wait_asynccnt(0);
#else
  asm volatile("s_wait_asynccnt 0" ::: "memory");
#endif
#endif
}

// ---------- bf16 helpers ----------
__device__ __forceinline__ unsigned short f32_to_bf16(float f) {
  union { float f; unsigned u; } c; c.f = f;
  unsigned u = c.u;
  return (unsigned short)((u + 0x7FFFu + ((u >> 16) & 1u)) >> 16);
}
__device__ __forceinline__ float bf16_to_f32(unsigned short b) {
  union { unsigned u; float f; } c; c.u = ((unsigned)b) << 16;
  return c.f;
}

// ---------- fp32 -> bf16 conversion ----------
__global__ __launch_bounds__(256) void cvt_f32_bf16(const float* __restrict__ src,
                                                    unsigned short* __restrict__ dst,
                                                    int n) {
  int i = blockIdx.x * blockDim.x + threadIdx.x;
  if (i < n) dst[i] = f32_to_bf16(src[i]);
}

// ---------- WMMA GEMM: C[M,N] = scale * A[M,K] * B^T, B stored N-major ----------
// Block: 256 threads = 8 waves (4 along M x 2 along N). Block tile 128x128.
// Wave tile 32x64: 2 A-frags x 4 B-frags -> 8 f32 accumulators, 8 WMMA per k-block.
// A/B k-tiles double-buffered in LDS via async copies.
union FragU { v16bf v; uint4 q[2]; };

template <bool OUT_BF16, bool TRANS_OUT>
__global__ __launch_bounds__(256, 1)
void wmma_gemm_bf16(const unsigned short* __restrict__ A,
                    const unsigned short* __restrict__ B,
                    void* __restrict__ Cp,
                    int K, int lda, int ldb, int ldc, float scale) {
  __shared__ unsigned short sA[2][BM * LST];
  __shared__ unsigned short sB[2][BN * LST];

  const int t    = threadIdx.x;
  const int lane = t & 31;
  const int wave = t >> 5;
  const int wm   = wave & 3;   // 0..3 along M (32 rows each)
  const int wn   = wave >> 2;  // 0..1 along N (64 cols each)
  const int half = lane >> 4;
  const int l16  = lane & 15;

  const int rowBase = blockIdx.y * BM;
  const int colBase = blockIdx.x * BN;

  // tile-copy mapping: 512 16B-chunks per tile, 2 per thread
  // chunk c: row = c/4, chunk-in-row = c%4 (KB=32 -> 4 chunks of 8 bf16)
  const int c0 = t, c1 = t + 256;

  v8f acc[2][4] = {};

  const int nKB = K / KB;

  // ---- prologue: stage k-block 0 into buffer 0 ----
  {
    const int r0 = c0 >> 2, ch0 = c0 & 3, r1 = c1 >> 2, ch1 = c1 & 3;
    cp16_g2l(&sA[0][r0 * LST + ch0 * 8], A + (size_t)(rowBase + r0) * lda + ch0 * 8);
    cp16_g2l(&sA[0][r1 * LST + ch1 * 8], A + (size_t)(rowBase + r1) * lda + ch1 * 8);
    cp16_g2l(&sB[0][r0 * LST + ch0 * 8], B + (size_t)(colBase + r0) * ldb + ch0 * 8);
    cp16_g2l(&sB[0][r1 * LST + ch1 * 8], B + (size_t)(colBase + r1) * ldb + ch1 * 8);
  }

  for (int kt = 0; kt < nKB; ++kt) {
    wait_async_copies();
    __syncthreads();               // k-tile kt resident in buf kt&1, safe to read

    // ---- prefetch k-block kt+1 into the other buffer (overlaps WMMA below) ----
    if (kt + 1 < nKB) {
      const int kb = (kt + 1) * KB;
      const int b = (kt + 1) & 1;
      const int r0 = c0 >> 2, ch0 = c0 & 3, r1 = c1 >> 2, ch1 = c1 & 3;
      cp16_g2l(&sA[b][r0 * LST + ch0 * 8], A + (size_t)(rowBase + r0) * lda + kb + ch0 * 8);
      cp16_g2l(&sA[b][r1 * LST + ch1 * 8], A + (size_t)(rowBase + r1) * lda + kb + ch1 * 8);
      cp16_g2l(&sB[b][r0 * LST + ch0 * 8], B + (size_t)(colBase + r0) * ldb + kb + ch0 * 8);
      cp16_g2l(&sB[b][r1 * LST + ch1 * 8], B + (size_t)(colBase + r1) * ldb + kb + ch1 * 8);
    }

    // ---- fragment loads from LDS + 8 WMMAs ----
    const unsigned short* bufA = sA[kt & 1];
    const unsigned short* bufB = sB[kt & 1];

    FragU af[2];
#pragma unroll
    for (int sub = 0; sub < 2; ++sub) {
      // ISA 16-bit A layout: VGPR0-3 <- K = half*8+0..7 ; VGPR4-7 <- K = 16+half*8+0..7
      const unsigned short* ar = bufA + (wm * 32 + sub * 16 + l16) * LST;
      af[sub].q[0] = *reinterpret_cast<const uint4*>(ar + half * 8);
      af[sub].q[1] = *reinterpret_cast<const uint4*>(ar + 16 + half * 8);
    }
    FragU bf[4];
#pragma unroll
    for (int j = 0; j < 4; ++j) {
      // ISA 16-bit B layout: VGPR0-3 <- K = half*8+0..7 ; VGPR4-7 <- K = 16+half*8+0..7
      const unsigned short* br = bufB + (wn * 64 + j * 16 + l16) * LST + half * 8;
      bf[j].q[0] = *reinterpret_cast<const uint4*>(br);
      bf[j].q[1] = *reinterpret_cast<const uint4*>(br + 16);
    }
#pragma unroll
    for (int sub = 0; sub < 2; ++sub)
#pragma unroll
      for (int j = 0; j < 4; ++j)
        acc[sub][j] = __builtin_amdgcn_wmma_f32_16x16x32_bf16(
            false, af[sub].v, false, bf[j].v, (short)0, acc[sub][j], false, false);

    __syncthreads();               // all waves done reading buf kt&1 before it is reused
  }

  // ---- store: C(m = rowBase + wm*32 + sub*16 + half*8 + r, n = colBase + wn*64 + j*16 + l16)
#pragma unroll
  for (int sub = 0; sub < 2; ++sub) {
#pragma unroll
    for (int j = 0; j < 4; ++j) {
#pragma unroll
      for (int r = 0; r < 8; ++r) {
        const int m = rowBase + wm * 32 + sub * 16 + half * 8 + r;
        const int n = colBase + wn * 64 + j * 16 + l16;
        const float v = acc[sub][j][r] * scale;
        if (OUT_BF16) {
          unsigned short* C = (unsigned short*)Cp;
          if (TRANS_OUT) C[(size_t)n * ldc + m] = f32_to_bf16(v);
          else           C[(size_t)m * ldc + n] = f32_to_bf16(v);
        } else {
          float* C = (float*)Cp;
          if (TRANS_OUT) C[(size_t)n * ldc + m] = v;
          else           C[(size_t)m * ldc + n] = v;
        }
      }
    }
  }
}

// ---------- row softmax (in place, bf16), one block per row ----------
__global__ __launch_bounds__(256, 1)
void softmax_rows_bf16(unsigned short* __restrict__ S, int n /*=4096*/) {
  __shared__ float red[256];
  unsigned short* p = S + (size_t)blockIdx.x * n;
  const int t = threadIdx.x;

  float vals[16];
  float lmax = -3.0e38f;
#pragma unroll
  for (int i = 0; i < 16; ++i) {
    vals[i] = bf16_to_f32(p[t + i * 256]);
    lmax = fmaxf(lmax, vals[i]);
  }
  red[t] = lmax;
  __syncthreads();
  for (int s = 128; s > 0; s >>= 1) {
    if (t < s) red[t] = fmaxf(red[t], red[t + s]);
    __syncthreads();
  }
  const float rmax = red[0];
  __syncthreads();

  float lsum = 0.f;
#pragma unroll
  for (int i = 0; i < 16; ++i) {
    vals[i] = __expf(vals[i] - rmax);
    lsum += vals[i];
  }
  red[t] = lsum;
  __syncthreads();
  for (int s = 128; s > 0; s >>= 1) {
    if (t < s) red[t] += red[t + s];
    __syncthreads();
  }
  const float inv = 1.0f / red[0];
#pragma unroll
  for (int i = 0; i < 16; ++i)
    p[t + i * 256] = f32_to_bf16(vals[i] * inv);
}

// ---------- launch ----------
extern "C" void kernel_launch(void* const* d_in, const int* in_sizes, int n_in,
                              void* d_out, int out_size, void* d_ws, size_t ws_size,
                              hipStream_t stream) {
  const float* X  = (const float*)d_in[0];  // (4096, 1024)
  const float* WQ = (const float*)d_in[1];  // (1024, 1024)
  const float* WK = (const float*)d_in[2];
  const float* WV = (const float*)d_in[3];

  // workspace layout (bf16 elements)
  unsigned short* Xb = (unsigned short*)d_ws;                 // 4096x1024
  unsigned short* Wq = Xb + (size_t)SEQ * EMB;                // 1024x1024
  unsigned short* Wk = Wq + (size_t)DOUT * EMB;
  unsigned short* Wv = Wk + (size_t)DOUT * EMB;
  unsigned short* Qb = Wv + (size_t)DOUT * EMB;               // 4096x1024
  unsigned short* Kb = Qb + (size_t)SEQ * DOUT;               // 4096x1024
  unsigned short* Vt = Kb + (size_t)SEQ * DOUT;               // 1024x4096 (transposed V)
  unsigned short* Sm = Vt + (size_t)DOUT * SEQ;               // 4096x4096 scores/probs

  // 1) convert inputs to bf16
  cvt_f32_bf16<<<(SEQ * EMB) / 256, 256, 0, stream>>>(X, Xb, SEQ * EMB);
  cvt_f32_bf16<<<(DOUT * EMB) / 256, 256, 0, stream>>>(WQ, Wq, DOUT * EMB);
  cvt_f32_bf16<<<(DOUT * EMB) / 256, 256, 0, stream>>>(WK, Wk, DOUT * EMB);
  cvt_f32_bf16<<<(DOUT * EMB) / 256, 256, 0, stream>>>(WV, Wv, DOUT * EMB);

  // 2) projections: Q,K row-major bf16; V stored transposed (D x S)
  {
    dim3 grid(DOUT / BN, SEQ / BM);
    wmma_gemm_bf16<true, false><<<grid, 256, 0, stream>>>(Xb, Wq, Qb, EMB, EMB, EMB, DOUT, 1.0f);
    wmma_gemm_bf16<true, false><<<grid, 256, 0, stream>>>(Xb, Wk, Kb, EMB, EMB, EMB, DOUT, 1.0f);
    wmma_gemm_bf16<true, true ><<<grid, 256, 0, stream>>>(Xb, Wv, Vt, EMB, EMB, EMB, SEQ, 1.0f);
  }

  // 3) scores S = Q @ K^T / sqrt(1024)
  {
    dim3 grid(SEQ / BN, SEQ / BM);
    wmma_gemm_bf16<true, false><<<grid, 256, 0, stream>>>(Qb, Kb, Sm, EMB, EMB, EMB, SEQ,
                                                          1.0f / 32.0f);
  }

  // 4) softmax rows in place
  softmax_rows_bf16<<<SEQ, 256, 0, stream>>>(Sm, SEQ);

  // 5) O = P @ V  (B operand = V^T rows, contiguous over s)
  {
    dim3 grid(DOUT / BN, SEQ / BM);
    wmma_gemm_bf16<false, false><<<grid, 256, 0, stream>>>(Sm, Vt, d_out, SEQ, SEQ, SEQ, DOUT,
                                                           1.0f);
  }
}
